// ConvVAE_40003325395697
// MI455X (gfx1250) — compile-verified, matrix operand order checked by reading, not run on hardware
//
#include <hip/hip_runtime.h>
#include <hip/hip_bf16.h>
#include <math.h>

typedef _Float16 h16;
typedef __attribute__((ext_vector_type(16))) _Float16 v16h;
typedef __attribute__((ext_vector_type(8)))  float    v8f;

#define BATCH 128
#define LDIM  16
#define NEMB  512

// ---------------------------------------------------------------------------
// utility kernels
// ---------------------------------------------------------------------------
__global__ void k_f32_to_f16(const float* __restrict__ s, h16* __restrict__ d, long n) {
  long i = (long)blockIdx.x * blockDim.x + threadIdx.x;
  if (i < n) d[i] = (h16)s[i];
}

// NCHW f32 -> CBHW f16 (channel-outermost layout for the implicit-GEMM convs)
__global__ void k_nchw_to_cbhw_f16(const float* __restrict__ x, h16* __restrict__ y,
                                   int C, int HW2) {
  long i = (long)blockIdx.x * blockDim.x + threadIdx.x;
  long tot = (long)BATCH * C * HW2;
  if (i >= tot) return;
  int p = (int)(i % HW2);
  int c = (int)((i / HW2) % C);
  int b = (int)(i / ((long)HW2 * C));
  y[((long)c * BATCH + b) * HW2 + p] = (h16)x[i];
}

// conv weight (Cout,Cin,4,4)=[Cout,Kreal] f32 -> padded [Mp,Kp] f16, zero filled
__global__ void k_pack_w(const float* __restrict__ w, h16* __restrict__ wp,
                         int Cout, int Kreal, int Mp, int Kp) {
  long i = (long)blockIdx.x * blockDim.x + threadIdx.x;
  long tot = (long)Mp * Kp;
  if (i >= tot) return;
  int m = (int)(i / Kp);
  int k = (int)(i % Kp);
  wp[i] = (m < Cout && k < Kreal) ? (h16)w[(long)m * Kreal + k] : (h16)0.f;
}

// transposed-conv weight (Cin,Cout,4,4) f32 -> padded [Mp,Kp] f16 ([co, ci*16+r])
__global__ void k_pack_wT(const float* __restrict__ w, h16* __restrict__ wp,
                          int Cout, int Kreal, int Mp, int Kp) {
  long i = (long)blockIdx.x * blockDim.x + threadIdx.x;
  long tot = (long)Mp * Kp;
  if (i >= tot) return;
  int m = (int)(i / Kp);
  int k = (int)(i % Kp);
  h16 v = (h16)0.f;
  if (m < Cout && k < Kreal) {
    int ci = k >> 4, r = k & 15;
    v = (h16)w[((long)ci * Cout + m) * 16 + r];
  }
  wp[i] = v;
}

// dense weight W[K,N] f32 -> Bt[N,K] f16 (fragment pairs become dword loads)
__global__ void k_pack_dense(const float* __restrict__ W, h16* __restrict__ Bt,
                             int K, int N) {
  long i = (long)blockIdx.x * blockDim.x + threadIdx.x;
  if (i >= (long)K * N) return;
  int n = (int)(i / K);
  int k = (int)(i % K);
  Bt[i] = (h16)W[(long)k * N + n];
}

// branch-free patch gather: clamped address + cndmask zero select
template <int TR>
__device__ __forceinline__ h16 gatherB(const h16* __restrict__ act, int IH, int IW,
                                       long planeB, int nb, int oh, int ow,
                                       int ci, int kh, int kw, int pad, bool vn) {
  int ih, iw; bool ok;
  if (TR == 0) {
    ih = oh * 2 - pad + kh; iw = ow * 2 - pad + kw;
    ok = (ih >= 0) & (ih < IH) & (iw >= 0) & (iw < IW);
  } else {
    int th = oh + pad - kh, tw = ow + pad - kw;
    ok = (th >= 0) & (tw >= 0) & ((th & 1) == 0) & ((tw & 1) == 0);
    ih = th >> 1; iw = tw >> 1;
    ok = ok & (ih < IH) & (iw < IW);
  }
  ok = ok & vn;
  int ihc = min(max(ih, 0), IH - 1);
  int iwc = min(max(iw, 0), IW - 1);
  h16 t = act[((long)ci * BATCH + nb) * planeB + (long)ihc * IW + iwc];
  return ok ? t : (h16)0.f;
}

// ---------------------------------------------------------------------------
// Implicit-GEMM conv / transposed conv via WMMA.
//   out[Cout, P] = Wpad[Mp, Kp] x Patch[Kp, P] + bias
// One wave32 owns a 32x32 tile (4 accumulators, 4 wmma per K-step);
// 4 waves/block share the 32x32 weight tile staged in LDS via b128 loads.
// ---------------------------------------------------------------------------
template <int TR>
__global__ void k_conv_gemm(const h16* __restrict__ wpad, const h16* __restrict__ act,
                            const float* __restrict__ bias, float* __restrict__ out,
                            int Cout, int Kp, int IH, int IW, int OH, int OW, int pad)
{
  const int P    = BATCH * OH * OW;
  const int lane = threadIdx.x & 31;
  const int wv   = threadIdx.x >> 5;
  const int m0   = blockIdx.y * 32;
  const int n0   = (blockIdx.x * 4 + wv) * 32;
  const int mrow = lane & 15;
  const int grp  = lane >> 4;
  const long planeB = (long)IH * IW;

  __shared__ __align__(16) h16 lA[32 * 32];

  const int nA = n0 + mrow;
  const int nB = n0 + 16 + mrow;
  const bool vn0 = (nA < P), vn1 = (nB < P);
  int na = min(nA, P - 1), nbx = min(nB, P - 1);
  int nb0 = na / (OH * OW); int r0 = na % (OH * OW); int oh0 = r0 / OW, ow0 = r0 % OW;
  int nb1 = nbx / (OH * OW); int r1 = nbx % (OH * OW); int oh1 = r1 / OW, ow1 = r1 % OW;

  v8f a00 = {}, a01 = {}, a10 = {}, a11 = {};

  for (int kk = 0; kk < Kp; kk += 32) {
    // stage 32x32 weight tile: unconditional b128 per thread
    {
      int t   = threadIdx.x;
      int row = t >> 2;
      int col = (t & 3) * 8;
      const h16* pA = wpad + (long)(m0 + row) * Kp + kk + col;
      uint4 w4 = *(const uint4*)pA;
      __syncthreads();
      *(uint4*)(lA + row * 32 + col) = w4;
      __syncthreads();
      __builtin_prefetch(pA + 32, 0, 1);
    }

    v16h aLo, aHi, b0, b1;
#pragma unroll
    for (int v = 0; v < 8; ++v) {
      int kb = (v < 4) ? (8 * grp + 2 * v) : (16 + 8 * grp + 2 * (v - 4));
      aLo[2 * v]     = lA[mrow * 32 + kb];
      aLo[2 * v + 1] = lA[mrow * 32 + kb + 1];
      aHi[2 * v]     = lA[(mrow + 16) * 32 + kb];
      aHi[2 * v + 1] = lA[(mrow + 16) * 32 + kb + 1];
      int k  = kk + kb;
      int ci = k >> 4, r = k & 15, kh = r >> 2, kw = r & 3;  // kw even
      b0[2 * v]     = gatherB<TR>(act, IH, IW, planeB, nb0, oh0, ow0, ci, kh, kw,     pad, vn0);
      b0[2 * v + 1] = gatherB<TR>(act, IH, IW, planeB, nb0, oh0, ow0, ci, kh, kw + 1, pad, vn0);
      b1[2 * v]     = gatherB<TR>(act, IH, IW, planeB, nb1, oh1, ow1, ci, kh, kw,     pad, vn1);
      b1[2 * v + 1] = gatherB<TR>(act, IH, IW, planeB, nb1, oh1, ow1, ci, kh, kw + 1, pad, vn1);
    }
    a00 = __builtin_amdgcn_wmma_f32_16x16x32_f16(false, aLo, false, b0, (short)0, a00, false, false);
    a01 = __builtin_amdgcn_wmma_f32_16x16x32_f16(false, aLo, false, b1, (short)0, a01, false, false);
    a10 = __builtin_amdgcn_wmma_f32_16x16x32_f16(false, aHi, false, b0, (short)0, a10, false, false);
    a11 = __builtin_amdgcn_wmma_f32_16x16x32_f16(false, aHi, false, b1, (short)0, a11, false, false);
  }

#pragma unroll
  for (int r = 0; r < 8; ++r) {
    int row = r + 8 * grp;
    int miL = m0 + row, miH = m0 + 16 + row;
    if (miL < Cout) {
      float bL = bias[miL];
      if (vn0) out[(long)miL * P + nA] = a00[r] + bL;
      if (vn1) out[(long)miL * P + nB] = a01[r] + bL;
    }
    if (miH < Cout) {
      float bH = bias[miH];
      if (vn0) out[(long)miH * P + nA] = a10[r] + bH;
      if (vn1) out[(long)miH * P + nB] = a11[r] + bH;
    }
  }
}

// ---------------------------------------------------------------------------
// Dense GEMM via WMMA: C[M,N] = A[M,K] x Bt[N,K]^T + bias[N]; M % 32 == 0
// ---------------------------------------------------------------------------
__global__ void k_gemm(const h16* __restrict__ A, const h16* __restrict__ Bt,
                       const float* __restrict__ bias, float* __restrict__ C,
                       h16* __restrict__ C16, int M, int N, int K, int actm)
{
  const int lane = threadIdx.x & 31;
  const int wv   = threadIdx.x >> 5;
  const int m0   = blockIdx.y * 32;
  const int n0   = (blockIdx.x * 4 + wv) * 32;
  const int mrow = lane & 15;
  const int grp  = lane >> 4;
  __shared__ __align__(16) h16 lA[32 * 32];

  const int nA = n0 + mrow;
  const int nB = n0 + 16 + mrow;
  const bool vn0 = (nA < N), vn1 = (nB < N);
  const h16* pB0 = Bt + (long)min(nA, N - 1) * K;
  const h16* pB1 = Bt + (long)min(nB, N - 1) * K;

  v8f a00 = {}, a01 = {}, a10 = {}, a11 = {};

  for (int kk = 0; kk < K; kk += 32) {
    {
      int t   = threadIdx.x;
      int row = t >> 2;
      int col = (t & 3) * 8;
      const h16* pA = A + (long)(m0 + row) * K + kk + col;
      uint4 w4 = *(const uint4*)pA;
      __syncthreads();
      *(uint4*)(lA + row * 32 + col) = w4;
      __syncthreads();
      __builtin_prefetch(pA + 32, 0, 1);
    }
    v16h aLo, aHi, b0, b1;
#pragma unroll
    for (int v = 0; v < 8; ++v) {
      int kb = (v < 4) ? (8 * grp + 2 * v) : (16 + 8 * grp + 2 * (v - 4));
      aLo[2 * v]     = lA[mrow * 32 + kb];
      aLo[2 * v + 1] = lA[mrow * 32 + kb + 1];
      aHi[2 * v]     = lA[(mrow + 16) * 32 + kb];
      aHi[2 * v + 1] = lA[(mrow + 16) * 32 + kb + 1];
      union { unsigned u; h16 h[2]; } c0, c1;
      c0.u = *(const unsigned*)(pB0 + kk + kb);
      c1.u = *(const unsigned*)(pB1 + kk + kb);
      b0[2 * v] = c0.h[0]; b0[2 * v + 1] = c0.h[1];
      b1[2 * v] = c1.h[0]; b1[2 * v + 1] = c1.h[1];
    }
    a00 = __builtin_amdgcn_wmma_f32_16x16x32_f16(false, aLo, false, b0, (short)0, a00, false, false);
    a01 = __builtin_amdgcn_wmma_f32_16x16x32_f16(false, aLo, false, b1, (short)0, a01, false, false);
    a10 = __builtin_amdgcn_wmma_f32_16x16x32_f16(false, aHi, false, b0, (short)0, a10, false, false);
    a11 = __builtin_amdgcn_wmma_f32_16x16x32_f16(false, aHi, false, b1, (short)0, a11, false, false);
  }

#pragma unroll
  for (int r = 0; r < 8; ++r) {
    int row = r + 8 * grp;
    int miL = m0 + row, miH = m0 + 16 + row;
    float vv;
    if (vn0) {
      float bA = bias[nA];
      vv = a00[r] + bA;
      if (actm == 1) vv = (vv >= 0.f) ? vv : 0.01f * vv;
      C[(long)miL * N + nA] = vv;
      if (C16) C16[(long)miL * N + nA] = (h16)vv;
      vv = a10[r] + bA;
      if (actm == 1) vv = (vv >= 0.f) ? vv : 0.01f * vv;
      C[(long)miH * N + nA] = vv;
      if (C16) C16[(long)miH * N + nA] = (h16)vv;
    }
    if (vn1) {
      float bB = bias[nB];
      vv = a01[r] + bB;
      if (actm == 1) vv = (vv >= 0.f) ? vv : 0.01f * vv;
      C[(long)miL * N + nB] = vv;
      if (C16) C16[(long)miL * N + nB] = (h16)vv;
      vv = a11[r] + bB;
      if (actm == 1) vv = (vv >= 0.f) ? vv : 0.01f * vv;
      C[(long)miH * N + nB] = vv;
      if (C16) C16[(long)miH * N + nB] = (h16)vv;
    }
  }
}

// ---------------------------------------------------------------------------
// BatchNorm: per-channel stats over P, then normalize (+leaky/tanh), emit f16
// ---------------------------------------------------------------------------
__global__ void k_bn_stats(const float* __restrict__ y, float* __restrict__ mean,
                           float* __restrict__ var, long P) {
  int c = blockIdx.x;
  const float* p = y + (long)c * P;
  float s = 0.f, s2 = 0.f;
  for (long i = threadIdx.x; i < P; i += blockDim.x) { float v = p[i]; s += v; s2 += v * v; }
  __shared__ float rs[256], rs2[256];
  rs[threadIdx.x] = s; rs2[threadIdx.x] = s2;
  __syncthreads();
  for (int o = 128; o > 0; o >>= 1) {
    if (threadIdx.x < o) { rs[threadIdx.x] += rs[threadIdx.x + o]; rs2[threadIdx.x] += rs2[threadIdx.x + o]; }
    __syncthreads();
  }
  if (threadIdx.x == 0) {
    float m = rs[0] / (float)P;
    mean[c] = m;
    var[c]  = rs2[0] / (float)P - m * m;
  }
}

__global__ void k_bn_apply(float* __restrict__ y, h16* __restrict__ y16,
                           const float* __restrict__ g, const float* __restrict__ be,
                           const float* __restrict__ mean, const float* __restrict__ var,
                           long P, long total, int actm) {
  long i = (long)blockIdx.x * blockDim.x + threadIdx.x;
  if (i >= total) return;
  int c = (int)(i / P);
  float v = y[i];
  v = g[c] * (v - mean[c]) * rsqrtf(var[c] + 1e-5f) + be[c];
  if (actm == 1)      v = (v >= 0.f) ? v : 0.01f * v;
  else if (actm == 2) v = tanhf(v);
  y[i] = v;
  if (y16) y16[i] = (h16)v;
}

// mean over spatial dims: h[D, B, HW2] -> lat[B, D]
__global__ void k_mean_hw(const float* __restrict__ h, float* __restrict__ lat, int HW2) {
  int i = blockIdx.x * blockDim.x + threadIdx.x;
  if (i >= BATCH * 512) return;
  int b = i / 512, d = i % 512;
  const float* p = h + ((long)d * BATCH + b) * HW2;
  float s = 0.f;
  for (int k = 0; k < HW2; ++k) s += p[k];
  lat[i] = s / (float)HW2;
}

__global__ void k_std(const float* __restrict__ lv, float* __restrict__ st, int n) {
  int i = blockIdx.x * blockDim.x + threadIdx.x;
  if (i < n) st[i] = expf(0.5f * lv[i]);
}

// cov_matrix / cov_tril_s / quant_input  (one block per batch element)
__global__ void k_latent_tail(const float* __restrict__ cor, const float* __restrict__ mu,
                              const float* __restrict__ eps, const float* __restrict__ stdb,
                              float* __restrict__ cov_out, float* __restrict__ quant) {
  int b = blockIdx.x;
  int t = threadIdx.x;
  int i = t >> 4, j = t & 15;
  __shared__ float ct[16][16];
  __shared__ float cts[16][16];
  float stdj  = stdb[b * 16 + j];
  float stdlj = stdb[(BATCH - 1) * 16 + j];
  float stdi  = stdb[b * 16 + i];
  float tril  = (j < i) ? cor[b * 256 + i * 16 + j] : 0.f;
  float ctij  = tril * stdlj * stdj;
  ct[i][j]  = ctij;
  cts[i][j] = ctij + ((i == j) ? stdi * stdi : 0.f);
  __syncthreads();
  cov_out[b * 256 + t] = ct[i][j] + ct[j][i] + ((i == j) ? stdi : 0.f);
  if (j == 0) {
    float s = mu[b * 16 + i];
    for (int jj = 0; jj < 16; ++jj) s += cts[i][jj] * eps[b * 16 + jj];
    quant[b * 16 + i] = s;
  }
}

// KL term from last batch element: 16x16 triangular solve (tiny, single thread)
__global__ void k_kl(const float* __restrict__ cov, const float* __restrict__ quant,
                     float* __restrict__ kl) {
  if (threadIdx.x != 0 || blockIdx.x != 0) return;
  float L[16][16], Ai[16][16];
  const float* cm = cov + (BATCH - 1) * 256;
  for (int i = 0; i < 16; ++i)
    for (int j = 0; j < 16; ++j) L[i][j] = (j <= i) ? cm[i * 16 + j] : 0.f;
  for (int col = 0; col < 16; ++col)
    for (int i = 0; i < 16; ++i) {
      float s = (i == col) ? 1.f : 0.f;
      for (int j = 0; j < i; ++j) s -= L[i][j] * Ai[j][col];
      Ai[i][col] = s / L[i][i];
    }
  float tr = 0.f;
  for (int i = 0; i < 16; ++i)
    for (int j = 0; j < 16; ++j) tr += Ai[i][j] * Ai[i][j];
  const float* m = quant + (BATCH - 1) * 16;
  float quad = 0.f;
  for (int i = 0; i < 16; ++i) {
    float s = 0.f;
    for (int j = 0; j < 16; ++j) s += Ai[i][j] * m[j];
    quad += s * s;
  }
  float logdet = 0.f;
  for (int i = 0; i < 16; ++i) logdet += logf(L[i][i]);
  logdet *= 2.f;
  kl[0] = 0.5f * (tr + quad - 16.f + logdet);
}

// VQ
__global__ void k_vq_dist(const float* __restrict__ q, const float* __restrict__ cb,
                          float* __restrict__ d) {
  int i = blockIdx.x * blockDim.x + threadIdx.x;
  if (i >= BATCH * NEMB) return;
  int b = i / NEMB, e = i % NEMB;
  float s = 0.f;
  for (int j = 0; j < 16; ++j) { float t = q[b * 16 + j] - cb[e * 16 + j]; s += t * t; }
  d[i] = s;
}

__global__ void k_argmin(const float* __restrict__ d, int* __restrict__ idx) {
  int b = blockIdx.x, t = threadIdx.x;
  float best = 3.4e38f; int bi = 0;
  for (int e = t; e < NEMB; e += 256) {
    float v = d[b * NEMB + e];
    if (v < best) { best = v; bi = e; }
  }
  __shared__ float sv[256]; __shared__ int si[256];
  sv[t] = best; si[t] = bi;
  __syncthreads();
  for (int o = 128; o > 0; o >>= 1) {
    if (t < o) {
      if (sv[t + o] < sv[t] || (sv[t + o] == sv[t] && si[t + o] < si[t])) {
        sv[t] = sv[t + o]; si[t] = si[t + o];
      }
    }
    __syncthreads();
  }
  if (t == 0) idx[b] = si[0];
}

__global__ void k_hist(const int* __restrict__ idx, float* __restrict__ hist) {
  int t = threadIdx.x;
  if (t < BATCH) atomicAdd(&hist[idx[t]], 1.0f);
}

__global__ void k_vq_losses(const float* __restrict__ q, const float* __restrict__ cb,
                            const int* __restrict__ idx, float* __restrict__ zq,
                            float* __restrict__ out) {
  int t = threadIdx.x;
  float s = 0.f;
  for (int i = t; i < BATCH * 16; i += 256) {
    int b = i >> 4, j = i & 15;
    float z = cb[idx[b] * 16 + j];
    zq[i] = z;
    float d = z - q[i];
    s += d * d;
  }
  __shared__ float r[256];
  r[t] = s;
  __syncthreads();
  for (int o = 128; o > 0; o >>= 1) { if (t < o) r[t] += r[t + o]; __syncthreads(); }
  if (t == 0) out[0] = 1.25f * r[0];   // (1 + BETA) * sum((zq - zf)^2)
}

// q = z_q @ post_q[0] + post_q[1], written as f16 decoder input [C=512, B]
__global__ void k_q_linear(const float* __restrict__ zq, const float* __restrict__ W,
                           const float* __restrict__ bias, h16* __restrict__ act) {
  int i = blockIdx.x * blockDim.x + threadIdx.x;
  if (i >= BATCH * 512) return;
  int b = i / 512, n = i % 512;
  float s = bias[n];
  for (int j = 0; j < 16; ++j) s += zq[b * 16 + j] * W[j * 512 + n];
  act[(long)n * BATCH + b] = (h16)s;
}

// sigmoid + CBHW -> NCHW for reconstruction
__global__ void k_sigmoid_nchw(const float* __restrict__ y, float* __restrict__ out,
                               int C, int HW2) {
  long i = (long)blockIdx.x * blockDim.x + threadIdx.x;
  long tot = (long)C * BATCH * HW2;
  if (i >= tot) return;
  int c = (int)(i / ((long)BATCH * HW2));
  long rem = i % ((long)BATCH * HW2);
  int b = (int)(rem / HW2);
  int p = (int)(rem % HW2);
  float v = y[i];
  v = 1.f / (1.f + expf(-v));
  out[((long)b * C + c) * HW2 + p] = v;
}

__global__ void k_finalize(const float* __restrict__ hist, const float* __restrict__ kl,
                           const float* __restrict__ embloss, float* __restrict__ out4) {
  if (threadIdx.x != 0 || blockIdx.x != 0) return;
  float s = 0.f;
  for (int e = 0; e < NEMB; ++e) {
    float em = hist[e] / (float)BATCH;
    s += em * logf(em + 1e-10f);
  }
  float perp = expf(-s);
  float bits = logf(perp);
  float bl = bits - kl[0];
  if (bl < 0.f) bl = 0.f;
  out4[0] = embloss[0];
  out4[1] = bits;
  out4[2] = perp;
  out4[3] = bl;
}

// ---------------------------------------------------------------------------
// host orchestration
// ---------------------------------------------------------------------------
extern "C" void kernel_launch(void* const* d_in, const int* in_sizes, int n_in,
                              void* d_out, int out_size, void* d_ws, size_t ws_size,
                              hipStream_t stream) {
  (void)in_sizes; (void)n_in; (void)out_size; (void)ws_size;
  const float* x   = (const float*)d_in[0];
  const float* eps = (const float*)d_in[1];

  size_t off = 0;
  auto alloc = [&](size_t bytes) {
    void* p = (char*)d_ws + off;
    off += (bytes + 255) & ~(size_t)255;
    return p;
  };
  h16*   xc16  = (h16*)alloc((size_t)4 * BATCH * 128 * 128 * sizeof(h16)); // 4th ch zero
  h16*   h1    = (h16*)alloc((size_t)16777216 * sizeof(h16));
  h16*   h2    = (h16*)alloc((size_t)16777216 * sizeof(h16));
  float* fbuf  = (float*)alloc((size_t)16777216 * sizeof(float));
  h16*   wpack = (h16*)alloc((size_t)2097152 * sizeof(h16));
  h16*   wd16  = (h16*)alloc((size_t)262144 * sizeof(h16));
  float* meanb = (float*)alloc(512 * 4);
  float* varb  = (float*)alloc(512 * 4);
  float* latf  = (float*)alloc(65536 * 4);
  h16*   lat16 = (h16*)alloc(65536 * 2);
  h16*   hidA  = (h16*)alloc(65536 * 2);
  h16*   hidB  = (h16*)alloc(65536 * 2);
  float* densef = (float*)alloc(65536 * 4);
  float* corf  = (float*)alloc(32768 * 4);
  float* logvf = (float*)alloc(2048 * 4);
  float* stdb  = (float*)alloc(2048 * 4);
  float* quantf = (float*)alloc(2048 * 4);
  float* zqf   = (float*)alloc(2048 * 4);
  float* dmat  = (float*)alloc(65536 * 4);
  int*   idxb  = (int*)alloc(128 * 4);
  float* hist  = (float*)alloc(512 * 4);
  float* klb   = (float*)alloc(256);
  float* embb  = (float*)alloc(256);

  float* out = (float*)d_out;
  const long RECON = (long)BATCH * 3 * 66 * 66;
  const long XN    = (long)BATCH * 3 * 128 * 128;
  float* out_recon = out;
  float* out_x     = out + RECON;
  float* out_mu    = out_x + XN;
  float* out_cov   = out_mu + BATCH * LDIM;
  float* out_scal  = out_cov + BATCH * LDIM * LDIM;

  // conv layer: pack padded weights, implicit-GEMM WMMA conv, optional BN+act
  auto conv = [&](const h16* ain, h16* aout, int CinEff, int Kreal, int Cout,
                  int IH, int OH, int pad, int transposed, int wi, int bni,
                  int actm, int do_bn) {
    int Kp = CinEff * 16;
    int Mp = (Cout + 31) & ~31;
    long wn = (long)Mp * Kp;
    if (!transposed)
      k_pack_w<<<(unsigned)((wn + 255) / 256), 256, 0, stream>>>(
          (const float*)d_in[wi], wpack, Cout, Kreal, Mp, Kp);
    else
      k_pack_wT<<<(unsigned)((wn + 255) / 256), 256, 0, stream>>>(
          (const float*)d_in[wi], wpack, Cout, Kreal, Mp, Kp);
    long P = (long)BATCH * OH * OH;
    dim3 gg((unsigned)((P + 127) / 128), (unsigned)(Mp / 32));
    if (!transposed)
      k_conv_gemm<0><<<gg, 128, 0, stream>>>(wpack, ain, (const float*)d_in[wi + 1],
                                             fbuf, Cout, Kp, IH, IH, OH, OH, pad);
    else
      k_conv_gemm<1><<<gg, 128, 0, stream>>>(wpack, ain, (const float*)d_in[wi + 1],
                                             fbuf, Cout, Kp, IH, IH, OH, OH, pad);
    if (do_bn) {
      k_bn_stats<<<Cout, 256, 0, stream>>>(fbuf, meanb, varb, P);
      long tot = (long)Cout * P;
      k_bn_apply<<<(unsigned)((tot + 255) / 256), 256, 0, stream>>>(
          fbuf, aout, (const float*)d_in[bni], (const float*)d_in[bni + 1],
          meanb, varb, P, tot, actm);
    }
  };

  auto gemm = [&](const h16* A, int wi, int bi, float* C, h16* C16,
                  int M, int N, int K, int actm) {
    long wn = (long)K * N;
    k_pack_dense<<<(unsigned)((wn + 255) / 256), 256, 0, stream>>>(
        (const float*)d_in[wi], wd16, K, N);
    dim3 gg((unsigned)((N + 127) / 128), (unsigned)(M / 32));
    k_gemm<<<gg, 128, 0, stream>>>(A, wd16, (const float*)d_in[bi], C, C16, M, N, K, actm);
  };

  // ---- input: zero 4-channel pad, then NCHW -> CBHW f16
  hipMemsetAsync(xc16, 0, (size_t)4 * BATCH * 128 * 128 * sizeof(h16), stream);
  k_nchw_to_cbhw_f16<<<(unsigned)((XN + 255) / 256), 256, 0, stream>>>(x, xc16, 3, 128 * 128);

  // ---- encoder
  conv(xc16, h1, 4, 48, 32, 128, 64, 1, 0, 2, 12, 1, 1);
  conv(h1, h2, 32, 512, 64, 64, 32, 1, 0, 4, 14, 1, 1);
  conv(h2, h1, 64, 1024, 128, 32, 16, 1, 0, 6, 16, 1, 1);
  conv(h1, h2, 128, 2048, 256, 16, 8, 1, 0, 8, 18, 1, 1);
  conv(h2, (h16*)nullptr, 256, 4096, 512, 8, 3, 0, 0, 10, 0, 0, 0);
  k_mean_hw<<<(BATCH * 512 + 255) / 256, 256, 0, stream>>>(fbuf, latf, 9);
  k_f32_to_f16<<<(65536 + 255) / 256, 256, 0, stream>>>(latf, lat16, 65536);
  gemm(lat16, 40, 41, densef, hidA, BATCH, 512, 512, 1);
  gemm(hidA, 42, 43, densef, hidB, BATCH, 512, 512, 1);
  gemm(hidB, 50, 51, out_mu, (h16*)nullptr, BATCH, 16, 512, 0);
  gemm(hidB, 52, 53, logvf, (h16*)nullptr, BATCH, 16, 512, 0);

  // ---- correlation branch
  conv(xc16, h1, 4, 48, 32, 128, 64, 1, 0, 20, 30, 1, 1);
  conv(h1, h2, 32, 512, 64, 64, 32, 1, 0, 22, 32, 1, 1);
  conv(h2, h1, 64, 1024, 128, 32, 16, 1, 0, 24, 34, 1, 1);
  conv(h1, h2, 128, 2048, 256, 16, 8, 1, 0, 26, 36, 1, 1);
  conv(h2, (h16*)nullptr, 256, 4096, 512, 8, 3, 0, 0, 28, 38, 2, 1); // BN + tanh
  k_mean_hw<<<(BATCH * 512 + 255) / 256, 256, 0, stream>>>(fbuf, latf, 9);
  k_f32_to_f16<<<(65536 + 255) / 256, 256, 0, stream>>>(latf, lat16, 65536);
  gemm(lat16, 44, 45, densef, hidA, BATCH, 512, 512, 1);
  gemm(hidA, 46, 47, densef, hidB, BATCH, 512, 512, 1);
  gemm(hidB, 48, 49, corf, (h16*)nullptr, BATCH, 256, 512, 0);

  // ---- latent tail + KL
  k_std<<<(2048 + 255) / 256, 256, 0, stream>>>(logvf, stdb, 2048);
  k_latent_tail<<<BATCH, 256, 0, stream>>>(corf, out_mu, eps, stdb, out_cov, quantf);
  k_kl<<<1, 32, 0, stream>>>(out_cov, quantf, klb);

  // ---- VQ
  const float* cb = (const float*)d_in[74];
  k_vq_dist<<<(BATCH * NEMB + 255) / 256, 256, 0, stream>>>(quantf, cb, dmat);
  k_argmin<<<BATCH, 256, 0, stream>>>(dmat, idxb);
  hipMemsetAsync(hist, 0, NEMB * sizeof(float), stream);
  k_hist<<<1, 128, 0, stream>>>(idxb, hist);
  k_vq_losses<<<1, 256, 0, stream>>>(quantf, cb, idxb, zqf, embb);
  k_q_linear<<<(BATCH * 512 + 255) / 256, 256, 0, stream>>>(
      zqf, (const float*)d_in[54], (const float*)d_in[55], h1);

  // ---- decoder (transposed convs)
  conv(h1, h2, 512, 8192, 256, 1, 4, 0, 1, 56, 66, 1, 1);
  conv(h2, h1, 256, 4096, 128, 4, 8, 1, 1, 58, 68, 1, 1);
  conv(h1, h2, 128, 2048, 64, 8, 16, 1, 1, 60, 70, 1, 1);
  conv(h2, h1, 64, 1024, 32, 16, 32, 1, 1, 62, 72, 1, 1);
  conv(h1, (h16*)nullptr, 32, 512, 3, 32, 66, 0, 1, 64, 0, 0, 0);
  k_sigmoid_nchw<<<(unsigned)((RECON + 255) / 256), 256, 0, stream>>>(fbuf, out_recon, 3, 66 * 66);

  // ---- scalars + x passthrough
  k_finalize<<<1, 32, 0, stream>>>(hist, klb, embb, out_scal);
  hipMemcpyAsync(out_x, x, XN * sizeof(float), hipMemcpyDeviceToDevice, stream);
}
